// CDXLSTM_32710470926848
// MI455X (gfx1250) — matrix operand order repaired and checked by reading, not compiled, not run on hardware
//
#include <hip/hip_runtime.h>
#include <hip/hip_bf16.h>
#include <cstdint>
#include <cstddef>
#include <cmath>

// ---------------------------------------------------------------------------
// Types / helpers
// ---------------------------------------------------------------------------
typedef __attribute__((ext_vector_type(16))) __bf16 v16bf;
typedef __attribute__((ext_vector_type(8)))  __bf16 v8bf;
typedef __attribute__((ext_vector_type(8)))  float  v8f;
typedef __attribute__((ext_vector_type(4)))  unsigned int v4u;
typedef __attribute__((ext_vector_type(8)))  int v8i;
typedef __attribute__((ext_vector_type(4)))  int v4i;

union HV { v8bf h[2]; v16bf v; };

static constexpr int   BBATCH = 4;
static constexpr int   NHH    = 4;
static constexpr float BNRC   = 0.9999950000374997f;   // 1/sqrt(1+1e-5)

#define DEVFN __device__ __forceinline__

DEVFN float  bf2f(__bf16 x) { return (float)x; }
DEVFN __bf16 f2bf(float x)  { return (__bf16)x; }
DEVFN float  sigm(float x)  { return 1.f / (1.f + expf(-x)); }
DEVFN float  siluf(float x) { return x * sigm(x); }
DEVFN float  geluf(float x) { return 0.5f * x * (1.f + erff(x * 0.70710678118654752f)); }
DEVFN float  logsigf(float x){ return (x >= 0.f) ? -log1pf(expf(-x)) : x - log1pf(expf(x)); }
DEVFN int    imin(int a, int b){ return a < b ? a : b; }

// ---------------------------------------------------------------------------
// Tensor Data Mover availability (this toolchain: 6-arg builtin,
// (uint32x4 g0, int32x8 g1, int32x4 g2, int32x4 g3, int32x8, i32 cpol))
// ---------------------------------------------------------------------------
#if defined(__has_builtin)
#if __has_builtin(__builtin_amdgcn_tensor_load_to_lds) && __has_builtin(__builtin_amdgcn_s_wait_tensorcnt)
#define HAS_TDM 1
#endif
#endif
#ifndef HAS_TDM
#define HAS_TDM 0
#endif

#if HAS_TDM
// Build D# groups for a 2-D bf16 tile load (tile: rows x cols elements, packed
// row-major into LDS).  OOB elements read as zero (free edge handling).
// Layout per CDNA5 ISA.
DEVFN void tdm_load_tile(unsigned ldsAddr, const __bf16* gsrc,
                         int cols /*tensor_dim0*/, int rows /*tensor_dim1*/,
                         long strideElems, int tile0, int tile1) {
  const unsigned long long ga = (unsigned long long)(uintptr_t)gsrc;
  v4u g0;
  g0[0] = 1u;                                            // count=1, user mode
  g0[1] = ldsAddr;                                       // lds byte address
  g0[2] = (unsigned)(ga & 0xffffffffull);                // global_addr[31:0]
  g0[3] = (unsigned)((ga >> 32) & 0x1ffffffull) | (2u << 30);  // ga[56:32]|type=2
  const unsigned td0 = (unsigned)cols, td1 = (unsigned)rows;
  const unsigned long long st0 = (unsigned long long)strideElems;
  v8i g1;
  g1[0] = (int)(1u << 16);                               // data_size=1 (2 bytes)
  g1[1] = (int)((td0 & 0xffffu) << 16);                  // tensor_dim0 lo
  g1[2] = (int)((td0 >> 16) | ((td1 & 0xffffu) << 16));  // dim0 hi | dim1 lo
  g1[3] = (int)((td1 >> 16) | ((unsigned)tile0 << 16));  // dim1 hi | tile_dim0
  g1[4] = (int)((unsigned)tile1);                        // tile_dim1 (tile2=0)
  g1[5] = (int)(unsigned)(st0 & 0xffffffffull);          // dim0_stride lo
  g1[6] = (int)(unsigned)((st0 >> 32) & 0xffffull);      // dim0_stride hi
  g1[7] = 0;
  v4i z4 = {0, 0, 0, 0};
  v8i z8 = {0, 0, 0, 0, 0, 0, 0, 0};
  __builtin_amdgcn_tensor_load_to_lds(g0, g1, z4, z4, z8, 0);
}
#endif

// ---------------------------------------------------------------------------
// GEMM: C[M,N] = epi( A[M,K](bf16) * Bt[N,K](bf16) ) (+ residual)
// block 128 threads = 4 waves, 64x64 tile, wave = 32x32 (2x2 wmma 16x16x32 bf16)
// TDMU=true: double-buffered LDS staged by the Tensor Data Mover.
// ---------------------------------------------------------------------------
enum {
  EPI_NONE = 0,
  EPI_BIAS,
  EPI_BIAS_GELU,
  EPI_BIAS_SIGMOID,
  EPI_SB_RELU,        // val*scale[n]+bias2[n], relu   (conv+BN+ReLU, folded)
  EPI_SB_RELU_SIG     // sigmoid(relu(val*scale+bias))
};

template <int EPI, bool OUT32, bool RES, bool TDMU>
__global__ __launch_bounds__(128) void k_gemm(
    const __bf16* __restrict__ A,  long sA, int lda,
    const __bf16* __restrict__ Bt, long sBt, int ldb,
    const float* __restrict__ scale, const float* __restrict__ bias,
    const __bf16* __restrict__ Res, long sRes, int ldr,
    void* __restrict__ Cp, long sC, int ldc,
    int M, int N, int K)
{
  __shared__ __bf16 As[2][64][32];
  __shared__ __bf16 Bs[2][64][32];
  const int b   = blockIdx.z;
  const __bf16* Ab = A  + (long)b * sA;
  const __bf16* Bb = Bt + (long)b * sBt;
  const int tid = threadIdx.x;
  const int m0 = blockIdx.y * 64, n0 = blockIdx.x * 64;
  const int wave = tid >> 5, lane = tid & 31;
  const int wm = (wave >> 1) * 32, wn = (wave & 1) * 32;
  const int ln = lane & 15, hi = lane >> 4;

  // vectorizable cooperative path (uniform checks)
  const bool vecA = ((lda & 7) == 0) && ((((unsigned long long)(uintptr_t)Ab) & 15ull) == 0);
  const bool vecB = ((ldb & 7) == 0) && ((((unsigned long long)(uintptr_t)Bb) & 15ull) == 0);
  auto coop_fill = [&](int k0) {
    #pragma unroll
    for (int it = 0; it < 2; ++it) {
      const int rr = tid + it * 128;            // 0..255 : 64 rows x 4 runs
      const int row = rr >> 2, kk = (rr & 3) * 8;
      const int gk = k0 + kk;
      { // A tile
        const int gm = m0 + row;
        v8bf val;
        if (gm < M && gk + 8 <= K && vecA) {
          val = *(const v8bf*)&Ab[(long)gm * lda + gk];
        } else {
          #pragma unroll
          for (int u = 0; u < 8; ++u)
            val[u] = (gm < M && gk + u < K) ? Ab[(long)gm * lda + gk + u] : (__bf16)0.f;
        }
        *(v8bf*)&As[0][row][kk] = val;
      }
      { // B tile
        const int gn = n0 + row;
        v8bf val;
        if (gn < N && gk + 8 <= K && vecB) {
          val = *(const v8bf*)&Bb[(long)gn * ldb + gk];
        } else {
          #pragma unroll
          for (int u = 0; u < 8; ++u)
            val[u] = (gn < N && gk + u < K) ? Bb[(long)gn * ldb + gk + u] : (__bf16)0.f;
        }
        *(v8bf*)&Bs[0][row][kk] = val;
      }
    }
    if (k0 + 32 < K) {
      const int pr = m0 + (tid & 63);
      if (pr < M) __builtin_prefetch(&Ab[(long)pr * lda + k0 + 32], 0, 1);
    }
  };

#if HAS_TDM
  auto tdm_issue = [&](int buf, int k0) {
    tdm_load_tile((unsigned)(unsigned long long)(uintptr_t)&As[buf][0][0],
                  Ab + (long)m0 * lda + k0, K - k0, M - m0, lda, 32, 64);
    tdm_load_tile((unsigned)(unsigned long long)(uintptr_t)&Bs[buf][0][0],
                  Bb + (long)n0 * ldb + k0, K - k0, N - n0, ldb, 32, 64);
  };
  if (TDMU && tid < 32) tdm_issue(0, 0);
#endif

  v8f acc[2][2] = {};
  for (int k0 = 0; k0 < K; k0 += 32) {
    int cur = 0;
#if HAS_TDM
    if (TDMU) {
      cur = (k0 >> 5) & 1;
      const bool more = (k0 + 32) < K;
      if (tid < 32) {
        if (more) {
          tdm_issue(cur ^ 1, k0 + 32);
          __builtin_amdgcn_s_wait_tensorcnt(2);   // retire current tile's 2 DMAs
        } else {
          __builtin_amdgcn_s_wait_tensorcnt(0);
        }
      }
    } else {
      coop_fill(k0);
    }
#else
    coop_fill(k0);
#endif
    __syncthreads();
    HV a[2], bbv[2];
    #pragma unroll
    for (int i = 0; i < 2; ++i) {
      const int row = wm + i * 16 + ln;
      a[i].h[0] = *(const v8bf*)&As[cur][row][hi * 8];
      a[i].h[1] = *(const v8bf*)&As[cur][row][hi * 8 + 16];
      const int col = wn + i * 16 + ln;
      bbv[i].h[0] = *(const v8bf*)&Bs[cur][col][hi * 8];
      bbv[i].h[1] = *(const v8bf*)&Bs[cur][col][hi * 8 + 16];
    }
    #pragma unroll
    for (int i = 0; i < 2; ++i)
      #pragma unroll
      for (int j = 0; j < 2; ++j)
        acc[i][j] = __builtin_amdgcn_wmma_f32_16x16x32_bf16(
            false, a[i].v, false, bbv[j].v, (short)0, acc[i][j], false, false);
    __syncthreads();
  }
  // epilogue
  #pragma unroll
  for (int i = 0; i < 2; ++i)
    #pragma unroll
    for (int j = 0; j < 2; ++j) {
      const int col = n0 + wn + j * 16 + ln;
      if (col >= N) continue;
      float s = 1.f, bv = 0.f;
      if (EPI == EPI_SB_RELU || EPI == EPI_SB_RELU_SIG) { s = scale[col]; bv = bias[col]; }
      else if (EPI != EPI_NONE) { bv = bias[col]; }
      #pragma unroll
      for (int r = 0; r < 8; ++r) {
        const int row = m0 + wm + i * 16 + r + hi * 8;
        if (row >= M) continue;
        float v = acc[i][j][r] * s + bv;
        if (EPI == EPI_SB_RELU)      v = fmaxf(v, 0.f);
        if (EPI == EPI_SB_RELU_SIG)  v = sigm(fmaxf(v, 0.f));
        if (EPI == EPI_BIAS_GELU)    v = geluf(v);
        if (EPI == EPI_BIAS_SIGMOID) v = sigm(v);
        if (RES) v += bf2f(Res[(long)b * sRes + (long)row * ldr + col]);
        if (OUT32) ((float*)Cp)[(long)b * sC + (long)row * ldc + col] = v;
        else ((__bf16*)Cp)[(long)b * sC + (long)row * ldc + col] = f2bf(v);
      }
    }
}

// ---------------------------------------------------------------------------
// Streaming stabilized mLSTM.  Q/K bf16 [B,NH,S,DHP]; V *transposed* bf16
// [B,NH,DHP,S]; cum/ig f32 [B,NH,S].  wave-per-16-rows, 32-key column pairs,
// bf16 WMMA for QK^T and P*V; only P (1KB/wave) lives in LDS.
// ---------------------------------------------------------------------------
template <int DHP>
__global__ __launch_bounds__(128) void k_mlstm(
    const __bf16* __restrict__ Q, const __bf16* __restrict__ Km,
    const __bf16* __restrict__ VT, const float* __restrict__ cum,
    const float* __restrict__ ig, float* __restrict__ H,
    int S, float scaleDH)
{
  constexpr int KC = DHP / 32;
  constexpr int OT = DHP / 16;
  __shared__ __bf16 Psh[4][16][32];
  const int wave = threadIdx.x >> 5, lane = threadIdx.x & 31;
  const int ln = lane & 15, hi = lane >> 4;
  const long head = (long)blockIdx.z * gridDim.y + blockIdx.y;
  const long base = head * (long)S;
  const __bf16* Qb = Q + base * DHP;
  const __bf16* Kb = Km + base * DHP;
  const __bf16* Vb = VT + head * (long)DHP * S;   // [DHP][S]
  const float* cm = cum + base;
  const float* ib = ig + base;
  float* Hb = H + base * DHP;
  const int r0 = blockIdx.x * 64 + wave * 16;
  if (r0 >= S) return;                 // no block-wide barriers used
  const int rEnd = imin(r0 + 15, S - 1);

  HV qa[KC];
  {
    const __bf16* qp = Qb + (long)imin(r0 + ln, S - 1) * DHP;
    #pragma unroll
    for (int c = 0; c < KC; ++c) {
      qa[c].h[0] = *(const v8bf*)(qp + c * 32 + hi * 8);
      qa[c].h[1] = *(const v8bf*)(qp + c * 32 + hi * 8 + 16);
    }
  }
  float cumRow[8], Mrun[8], Arun[8];
  #pragma unroll
  for (int r = 0; r < 8; ++r) {
    cumRow[r] = cm[imin(r0 + r + hi * 8, S - 1)];
    Mrun[r] = -1e30f; Arun[r] = 0.f;
  }
  v8f Oacc[OT] = {};
  __bf16 (*P)[32] = Psh[wave];

  for (int c0 = 0; c0 <= rEnd; c0 += 32) {
    float qk[2][8], ld[2][8];
    #pragma unroll
    for (int ct = 0; ct < 2; ++ct) {
      const int col = c0 + ct * 16 + ln;
      const int colC = imin(col, S - 1);
      const __bf16* kp = Kb + (long)colC * DHP;
      v8f sacc = {};
      #pragma unroll
      for (int c = 0; c < KC; ++c) {
        HV kf;
        kf.h[0] = *(const v8bf*)(kp + c * 32 + hi * 8);
        kf.h[1] = *(const v8bf*)(kp + c * 32 + hi * 8 + 16);
        sacc = __builtin_amdgcn_wmma_f32_16x16x32_bf16(
            false, qa[c].v, false, kf.v, (short)0, sacc, false, false);
      }
      const float cc = cm[colC], ii = ib[colC];
      #pragma unroll
      for (int r = 0; r < 8; ++r) {
        const int row = r0 + r + hi * 8;
        qk[ct][r] = sacc[r] * scaleDH;
        ld[ct][r] = (col <= row && col < S) ? (cumRow[r] - cc + ii) : -1e30f;
      }
    }
    float rho[8];
    #pragma unroll
    for (int r = 0; r < 8; ++r) {
      float mx = fmaxf(ld[0][r], ld[1][r]);
      #pragma unroll
      for (int o = 1; o < 16; o <<= 1) mx = fmaxf(mx, __shfl_xor(mx, o, 32));
      const float mnew = fmaxf(Mrun[r], mx);
      rho[r] = expf(Mrun[r] - mnew);
      Mrun[r] = mnew;
      const float p0 = qk[0][r] * expf(ld[0][r] - mnew);
      const float p1 = qk[1][r] * expf(ld[1][r] - mnew);
      P[r + hi * 8][ln]      = f2bf(p0);
      P[r + hi * 8][ln + 16] = f2bf(p1);
      float sm = p0 + p1;
      #pragma unroll
      for (int o = 1; o < 16; o <<= 1) sm += __shfl_xor(sm, o, 32);
      Arun[r] = Arun[r] * rho[r] + sm;
    }
    HV pa;
    pa.h[0] = *(const v8bf*)&P[ln][hi * 8];
    pa.h[1] = *(const v8bf*)&P[ln][hi * 8 + 16];
    #pragma unroll
    for (int t = 0; t < OT; ++t) {
      #pragma unroll
      for (int r = 0; r < 8; ++r) Oacc[t][r] *= rho[r];
      const __bf16* vp = Vb + (long)(t * 16 + ln) * S + c0;  // S%64==0 -> aligned
      HV vf;
      vf.h[0] = *(const v8bf*)(vp + hi * 8);
      vf.h[1] = *(const v8bf*)(vp + hi * 8 + 16);
      Oacc[t] = __builtin_amdgcn_wmma_f32_16x16x32_bf16(
          false, pa.v, false, vf.v, (short)0, Oacc[t], false, false);
    }
  }
  #pragma unroll
  for (int r = 0; r < 8; ++r) {
    const int row = r0 + r + hi * 8;
    if (row >= S) continue;
    const float nrm = fmaxf(fabsf(Arun[r]), expf(-Mrun[r])) + 1e-6f;
    #pragma unroll
    for (int t = 0; t < OT; ++t)
      Hb[(long)row * DHP + t * 16 + ln] = Oacc[t][r] / nrm;
  }
}

// ---------------------------------------------------------------------------
// Elementwise / layout kernels
// ---------------------------------------------------------------------------
#define IDX1D  long i = (long)blockIdx.x * 256 + threadIdx.x

__global__ void k_f32_bf16(const float* in, __bf16* out, long n) {
  IDX1D; if (i < n) out[i] = f2bf(in[i]);
}

// NCHW f32 -> [B,S,C] bf16 (optional diff)
__global__ void k_tobf_cl(const float* A, const float* Bn, __bf16* out,
                          int C, long S, int dodiff) {
  IDX1D; const long tot = (long)BBATCH * S * C; if (i >= tot) return;
  const int c = (int)(i % C); long t = i / C;
  const long s = t % S; const int b = (int)(t / S);
  const long src = ((long)b * C + c) * S + s;
  float v = A[src]; if (dodiff) v -= Bn[src];
  out[i] = f2bf(v);
}

__global__ void k_ln_rows(const __bf16* x, const float* g, __bf16* out, long R, int C) {
  IDX1D; if (i >= R) return;
  const __bf16* p = x + i * C;
  float mu = 0.f; for (int c = 0; c < C; ++c) mu += bf2f(p[c]); mu /= C;
  float var = 0.f;
  for (int c = 0; c < C; ++c) { float d = bf2f(p[c]) - mu; var += d * d; }
  var /= C;
  const float inv = rsqrtf(var + 1e-5f);
  __bf16* o = out + i * C;
  for (int c = 0; c < C; ++c) o[c] = f2bf((bf2f(p[c]) - mu) * inv * g[c]);
}

__global__ void k_conv4_silu(const __bf16* x, const float* w, const float* bi,
                             __bf16* out, int S, int I) {
  IDX1D; const long tot = (long)BBATCH * S * I; if (i >= tot) return;
  const int c = (int)(i % I); long t = i / I;
  const int s = (int)(t % S); const int b = (int)(t / S);
  float acc = bi[c];
  #pragma unroll
  for (int j = 0; j < 4; ++j) {
    const int sp = s - 3 + j;
    if (sp >= 0) acc += bf2f(x[((long)b * S + sp) * I + c]) * w[c * 4 + j];
  }
  out[i] = f2bf(siluf(acc));
}

__global__ void k_cumgates(const float* igf, const float* fgf,
                           float* cum, float* igT, int S) {
  if (threadIdx.x) return;
  const int n = blockIdx.x, b = blockIdx.y;
  const float* ip = igf + (long)b * S * NHH + n;
  const float* fp = fgf + (long)b * S * NHH + n;
  const long o = ((long)b * NHH + n) * S;
  float c = 0.f;
  for (int s = 0; s < S; ++s) {
    c += logsigf(fp[(long)s * NHH]);
    cum[o + s] = c;
    igT[o + s] = ip[(long)s * NHH];
  }
}

__global__ void k_pack_pad(const __bf16* x, __bf16* out, int S, int DH, int DHP) {
  IDX1D; const long tot = (long)BBATCH * NHH * S * DHP; if (i >= tot) return;
  const int dp = (int)(i % DHP); long t = i / DHP;
  const int s = (int)(t % S); t /= S;
  const int n = (int)(t % NHH); const int b = (int)(t / NHH);
  out[i] = (dp < DH) ? x[((long)b * S + s) * (NHH * DH) + n * DH + dp] : (__bf16)0.f;
}

// transposed pack: out [B,NH,DHP,S]
__global__ void k_pack_pad_T(const __bf16* x, __bf16* out, int S, int DH, int DHP) {
  IDX1D; const long tot = (long)BBATCH * NHH * DHP * S; if (i >= tot) return;
  const int s = (int)(i % S); long t = i / S;
  const int dp = (int)(t % DHP); t /= DHP;
  const int n = (int)(t % NHH); const int b = (int)(t / NHH);
  out[i] = (dp < DH) ? x[((long)b * S + s) * (NHH * DH) + n * DH + dp] : (__bf16)0.f;
}

__global__ void k_headln(const float* Hf, const float* g, __bf16* out,
                         int S, int DH, int DHP) {
  IDX1D; const long tot = (long)BBATCH * NHH * S; if (i >= tot) return;
  const int s = (int)(i % S); long t = i / S;
  const int n = (int)(t % NHH); const int b = (int)(t / NHH);
  const float* hp = Hf + i * DHP;
  float mu = 0.f; for (int d = 0; d < DH; ++d) mu += hp[d]; mu /= DH;
  float var = 0.f;
  for (int d = 0; d < DH; ++d) { float dd = hp[d] - mu; var += dd * dd; }
  var /= DH;
  const float inv = rsqrtf(var + 1e-5f);
  __bf16* op = out + ((long)b * S + s) * (NHH * DH) + n * DH;
  for (int d = 0; d < DH; ++d) op[d] = f2bf((hp[d] - mu) * inv * g[n * DH + d]);
}

__global__ void k_hs_gate(const __bf16* hn, const __bf16* ca, const __bf16* z,
                          const float* skip, __bf16* out, long R, int I) {
  IDX1D; const long tot = R * I; if (i >= tot) return;
  const int c = (int)(i % I);
  const float hs = bf2f(hn[i]) + skip[c] * bf2f(ca[i]);
  out[i] = f2bf(hs * siluf(bf2f(z[i])));
}

__global__ void k_cat2(const __bf16* a, const __bf16* b, __bf16* out,
                       long R, int C1, int C2) {
  IDX1D; const int C = C1 + C2; const long tot = R * C; if (i >= tot) return;
  const int c = (int)(i % C); const long r = i / C;
  out[i] = (c < C1) ? a[r * C1 + c] : b[r * C2 + (c - C1)];
}

__global__ void k_cat3(const __bf16* a, const __bf16* b, const __bf16* c,
                       __bf16* out, long R, int C) {
  IDX1D; const long tot = R * 3 * C; if (i >= tot) return;
  const int cc = (int)(i % (3 * C)); const long r = i / (3 * C);
  const __bf16* src = (cc < C) ? a : (cc < 2 * C ? b : c);
  out[i] = src[r * C + (cc % C)];
}

__global__ void k_dw3x3(const __bf16* x, const float* w, const float* bi,
                        __bf16* out, int H, int W, int C) {
  IDX1D; const long tot = (long)BBATCH * H * W * C; if (i >= tot) return;
  const int c = (int)(i % C); long t = i / C;
  const int xw = (int)(t % W); t /= W;
  const int y = (int)(t % H); const int b = (int)(t / H);
  float acc = bi[c];
  #pragma unroll
  for (int dy = -1; dy <= 1; ++dy)
    #pragma unroll
    for (int dx = -1; dx <= 1; ++dx) {
      const int yy = y + dy, xx = xw + dx;
      if (yy >= 0 && yy < H && xx >= 0 && xx < W)
        acc += bf2f(x[(((long)b * H + yy) * W + xx) * C + c]) *
               w[c * 9 + (dy + 1) * 3 + (dx + 1)];
    }
  out[i] = f2bf(acc);
}

__global__ void k_fold_bn(const float* g, const float* b, const float* pb,
                          float* s, float* b2, int C) {
  IDX1D; if (i >= C) return;
  const float sc = g[i] * BNRC;
  s[i] = sc;
  b2[i] = b[i] + (pb ? pb[i] * sc : 0.f);
}

__global__ void k_affine(const __bf16* x, const float* s, const float* b,
                         __bf16* out, long R, int C) {
  IDX1D; const long tot = R * C; if (i >= tot) return;
  const int c = (int)(i % C);
  out[i] = f2bf(bf2f(x[i]) * s[c] + b[c]);
}

__global__ void k_mean(const __bf16* x, __bf16* out, int H, int W, int C, int overW) {
  const int n = overW ? H : W, cnt = overW ? W : H;
  IDX1D; const long tot = (long)BBATCH * n * C; if (i >= tot) return;
  const int c = (int)(i % C); long t = i / C;
  const int a = (int)(t % n); const int b = (int)(t / n);
  float s = 0.f;
  for (int j = 0; j < cnt; ++j) {
    const int idx = overW ? (a * W + j) : (j * W + a);
    s += bf2f(x[((long)b * H * W + idx) * C + c]);
  }
  out[i] = f2bf(s / cnt);
}

__global__ void k_pos_add(__bf16* x, const float* pe, int S, int C) {
  IDX1D; const long tot = (long)BBATCH * S * C; if (i >= tot) return;
  const int c = (int)(i % C); long t = i / C; const int s = (int)(t % S);
  float p = (s + 0.5f) * 16.f / (float)S - 0.5f;
  p = fminf(fmaxf(p, 0.f), 15.f);
  const int i0 = (int)floorf(p), i1 = imin(i0 + 1, 15);
  const float fr = p - (float)i0;
  const float v = pe[c * 16 + i0] * (1.f - fr) + pe[c * 16 + i1] * fr;
  x[i] = f2bf(bf2f(x[i]) + v);
}

__global__ void k_comb3(const __bf16* d, const __bf16* xh, const __bf16* xw,
                        __bf16* out, int H, int W, int C) {
  IDX1D; const long tot = (long)BBATCH * H * W * C; if (i >= tot) return;
  const int c = (int)(i % C); long t = i / C;
  const int x = (int)(t % W); t /= W;
  const int y = (int)(t % H); const int b = (int)(t / H);
  out[i] = f2bf(bf2f(d[i]) + bf2f(xh[((long)b * H + y) * C + c]) +
                bf2f(xw[((long)b * W + x) * C + c]));
}

__global__ void k_gatemul(const __bf16* aw, const __bf16* xx, const __bf16* x,
                          __bf16* out, long S, int C, int use_xx) {
  IDX1D; const long tot = (long)BBATCH * S * C; if (i >= tot) return;
  const long r = i / C;
  float v = bf2f(aw[r]) * bf2f(x[i]);
  if (use_xx) v *= bf2f(xx[i]);
  out[i] = f2bf(v);
}

__global__ void k_flip(const __bf16* in, __bf16* out, long S, int C) {
  IDX1D; const long tot = (long)BBATCH * S * C; if (i >= tot) return;
  const int c = (int)(i % C); long t = i / C;
  const long s = t % S; const int b = (int)(t / S);
  out[i] = in[((long)b * S + (S - 1 - s)) * C + c];
}

__global__ void k_addflip(const __bf16* a, const __bf16* bsrc, __bf16* out,
                          long S, int C) {
  IDX1D; const long tot = (long)BBATCH * S * C; if (i >= tot) return;
  const int c = (int)(i % C); long t = i / C;
  const long s = t % S; const int b = (int)(t / S);
  out[i] = f2bf(bf2f(a[i]) + bf2f(bsrc[((long)b * S + (S - 1 - s)) * C + c]));
}

__global__ void k_add2(const __bf16* a, const __bf16* b, __bf16* out, long n) {
  IDX1D; if (i < n) out[i] = f2bf(bf2f(a[i]) + bf2f(b[i]));
}

__global__ void k_amp12(const __bf16* x, __bf16* out, int H, int W, int C) {
  IDX1D; const long tot = (long)BBATCH * 144 * C; if (i >= tot) return;
  const int c = (int)(i % C); long t = i / C;
  const int tx = (int)(t % 12); t /= 12;
  const int ty = (int)(t % 12); const int b = (int)(t / 12);
  const int ys = (ty * H) / 12, ye = ((ty + 1) * H + 11) / 12;
  const int xs = (tx * W) / 12, xe = ((tx + 1) * W + 11) / 12;
  float m = -3.4e38f;
  for (int y = ys; y < ye; ++y)
    for (int xw = xs; xw < xe; ++xw)
      m = fmaxf(m, bf2f(x[(((long)b * H + y) * W + xw) * C + c]));
  out[i] = f2bf(m);
}

__global__ void k_softmax_row(const float* in, __bf16* out, long R, int L, float sc) {
  IDX1D; if (i >= R) return;
  const float* p = in + i * L;
  float mx = -3.4e38f;
  for (int j = 0; j < L; ++j) mx = fmaxf(mx, p[j] * sc);
  float s = 0.f;
  for (int j = 0; j < L; ++j) s += expf(p[j] * sc - mx);
  const float inv = 1.f / s;
  __bf16* o = out + i * L;
  for (int j = 0; j < L; ++j) o[j] = f2bf(expf(p[j] * sc - mx) * inv);
}

// batched transpose (bf16): out[b][c][r] = in[b][r][off+c], in row-stride ldi
__global__ void k_transpose_bf(const __bf16* in, __bf16* out,
                               int R, int Cc, int ldi, int off) {
  IDX1D; const long tot = (long)BBATCH * Cc * R; if (i >= tot) return;
  const int r = (int)(i % R); long t = i / R;
  const int c = (int)(t % Cc); const int b = (int)(t / Cc);
  out[i] = in[((long)b * R + r) * ldi + off + c];
}

__global__ void k_bilinear(const __bf16* x, __bf16* out,
                           int Hi, int Wi, int Ho, int Wo, int C) {
  IDX1D; const long tot = (long)BBATCH * Ho * Wo * C; if (i >= tot) return;
  const int c = (int)(i % C); long t = i / C;
  const int xo = (int)(t % Wo); t /= Wo;
  const int yo = (int)(t % Ho); const int b = (int)(t / Ho);
  float fy = (yo + 0.5f) * (float)Hi / (float)Ho - 0.5f;
  float fx = (xo + 0.5f) * (float)Wi / (float)Wo - 0.5f;
  fy = fminf(fmaxf(fy, 0.f), (float)(Hi - 1));
  fx = fminf(fmaxf(fx, 0.f), (float)(Wi - 1));
  const int y0 = (int)floorf(fy), x0 = (int)floorf(fx);
  const int y1 = imin(y0 + 1, Hi - 1), x1 = imin(x0 + 1, Wi - 1);
  const float wy = fy - y0, wx = fx - x0;
  const long rb = (long)b * Hi * Wi;
  const float v00 = bf2f(x[(rb + (long)y0 * Wi + x0) * C + c]);
  const float v01 = bf2f(x[(rb + (long)y0 * Wi + x1) * C + c]);
  const float v10 = bf2f(x[(rb + (long)y1 * Wi + x0) * C + c]);
  const float v11 = bf2f(x[(rb + (long)y1 * Wi + x1) * C + c]);
  out[i] = f2bf((v00 * (1 - wx) + v01 * wx) * (1 - wy) + (v10 * (1 - wx) + v11 * wx) * wy);
}

__global__ void k_slc2nchw(const __bf16* in, float* out, int H, int W, int C) {
  IDX1D; const long tot = (long)BBATCH * C * H * W; if (i >= tot) return;
  const int xw = (int)(i % W); long t = i / W;
  const int y = (int)(t % H); t /= H;
  const int c = (int)(t % C); const int b = (int)(t / C);
  out[i] = bf2f(in[((long)b * H * W + (long)y * W + xw) * C + c]);
}

// ---------------------------------------------------------------------------
// Host side: arena, param cursor, module drivers
// ---------------------------------------------------------------------------
struct Arena {
  char* base; size_t cap, off;
  void* get(size_t bytes) {
    size_t a = (off + 255) & ~(size_t)255;
    if (a + bytes > cap) a = 0;          // compile-only fallback: wrap
    off = a + bytes;
    return base + a;
  }
  __bf16* bf(size_t n) { return (__bf16*)get(n * 2); }
  float* f32(size_t n) { return (float*)get(n * 4); }
};

static inline dim3 g1(long n) { return dim3((unsigned)((n + 255) / 256)); }

static __bf16* WB(Arena& ar, hipStream_t st, const float* w, long n) {
  __bf16* o = ar.bf((size_t)n);
  k_f32_bf16<<<g1(n), 256, 0, st>>>(w, o, n);
  return o;
}

template <int EPI, bool O32, bool RES, bool TDMU>
static void GEMM(hipStream_t st,
                 const __bf16* A, long sA, int lda,
                 const __bf16* Bt, long sBt, int ldb,
                 const float* sc, const float* bi,
                 const __bf16* res, long sRes, int ldr,
                 void* C, long sC, int ldc, int M, int N, int K) {
  dim3 gr((N + 63) / 64, (M + 63) / 64, BBATCH);
  k_gemm<EPI, O32, RES, TDMU><<<gr, 128, 0, st>>>(A, sA, lda, Bt, sBt, ldb, sc, bi,
                                                  res, sRes, ldr, C, sC, ldc, M, N, K);
}

// Param cursor (jax pytree alphabetical flatten order; blob or per-leaf)
struct PCur {
  void* const* din; int idx;
  const float* blob; size_t off; bool blobMode;
  const float* take(size_t n) {
    if (blobMode) { const float* p = blob + off; off += n; return p; }
    return (const float*)din[idx++];
  }
};
struct DsP  { const float *b,*dw,*dwb,*g,*pw,*pwb; };
struct C11P { const float *b,*g,*w; };
struct PcP  { const float *b,*w; };
struct MlpP { const float *b1,*b2,*w1,*w2; };
struct VilP { const float *cb,*cw,*down,*fgb,*fgw,*igb,*igw,*k,*ln_g,*mh_g,*q,*skip,*up,*v; };
struct AxialP { DsP cat,catA,catB; PcP convA,convB; const float *ph,*pw; VilP vh,vw; C11P xc; };
struct AttenP { DsP cat,catA,catB; PcP convA,convB; VilP v; };
struct LhP { const float *bn_b,*bn_g; C11P conv; PcP kv; MlpP mlp_h,mlp_l; const float *nb,*ng; };

static DsP takeDs(PCur& P, int cin, int cout) {
  DsP r; r.b=P.take(cout); r.dw=P.take((size_t)cin*9); r.dwb=P.take(cin);
  r.g=P.take(cout); r.pw=P.take((size_t)cout*cin); r.pwb=P.take(cout); return r;
}
static C11P takeC11(PCur& P, int cin, int cout) {
  C11P r; r.b=P.take(cout); r.g=P.take(cout); r.w=P.take((size_t)cout*cin); return r;
}
static PcP takePc(PCur& P, int cin, int cout) {
  PcP r; r.b=P.take(cout); r.w=P.take((size_t)cout*cin); return r;
}
static MlpP takeMlp(PCur& P, int cin, int cout) {
  MlpP r; r.b1=P.take(cin); r.b2=P.take(cout);
  r.w1=P.take((size_t)cin*cin); r.w2=P.take((size_t)cout*cin); return r;
}
static VilP takeVil(PCur& P, int dim) {
  const int I = 2*dim, DH = I/NHH; VilP r;
  r.cb=P.take(I); r.cw=P.take((size_t)I*4); r.down=P.take((size_t)dim*I);
  r.fgb=P.take(NHH); r.fgw=P.take((size_t)NHH*3*I);
  r.igb=P.take(NHH); r.igw=P.take((size_t)NHH*3*I);
  r.k=P.take((size_t)NHH*DH*DH); r.ln_g=P.take(dim); r.mh_g=P.take((size_t)NHH*DH);
  r.q=P.take((size_t)NHH*DH*DH); r.skip=P.take(I); r.up=P.take((size_t)2*I*dim);
  r.v=P.take((size_t)NHH*DH*DH); return r;
}
static AxialP takeAxial(PCur& P, int c) {
  AxialP r; r.cat=takeDs(P,2*c,c); r.catA=takeDs(P,2*c,c); r.catB=takeDs(P,2*c,c);
  r.convA=takePc(P,c,1); r.convB=takePc(P,c,1);
  r.ph=P.take((size_t)c*16); r.pw=P.take((size_t)c*16);
  r.vh=takeVil(P,c); r.vw=takeVil(P,c); r.xc=takeC11(P,c,c); return r;
}
static AttenP takeAtten(PCur& P, int c) {
  AttenP r; r.cat=takeDs(P,2*c,c); r.catA=takeDs(P,2*c,c); r.catB=takeDs(P,2*c,c);
  r.convA=takePc(P,c,1); r.convB=takePc(P,c,1); r.v=takeVil(P,c); return r;
}
static LhP takeLh(PCur& P, int cl, int ch) {
  LhP r; r.bn_b=P.take(cl); r.bn_g=P.take(cl);
  r.conv=takeC11(P,cl,ch); r.kv=takePc(P,cl,2*ch);
  r.mlp_h=takeMlp(P,ch,ch); r.mlp_l=takeMlp(P,cl,cl);
  r.nb=P.take(ch); r.ng=P.take(ch); return r;
}

// --------------------------- module drivers --------------------------------
static __bf16* run_vil(hipStream_t st, Arena& ar, const VilP& p,
                       const __bf16* x, int S, int dim) {
  const int I = 2*dim, DH = I/NHH, DHP = ((DH+31)/32)*32, K3 = 3*I;
  const long SD = (long)S*dim, SI = (long)S*I;

  __bf16* upW = WB(ar, st, p.up, (long)2*I*dim);
  __bf16* xln = ar.bf((size_t)BBATCH*SD);
  k_ln_rows<<<g1((long)BBATCH*S), 256, 0, st>>>(x, p.ln_g, xln, (long)BBATCH*S, dim);

  __bf16* xm = ar.bf((size_t)BBATCH*SI);
  __bf16* z  = ar.bf((size_t)BBATCH*SI);
  GEMM<EPI_NONE,false,false,true>(st, xln,SD,dim, upW,0,dim, 0,0, 0,0,0, xm,SI,I, S,I,dim);
  GEMM<EPI_NONE,false,false,true>(st, xln,SD,dim, upW+(long)I*dim,0,dim, 0,0, 0,0,0, z,SI,I, S,I,dim);

  __bf16* ca = ar.bf((size_t)BBATCH*SI);
  k_conv4_silu<<<g1((long)BBATCH*SI), 256, 0, st>>>(xm, p.cw, p.cb, ca, S, I);

  __bf16* qw = WB(ar, st, p.q, (long)NHH*DH*DH);
  __bf16* kw = WB(ar, st, p.k, (long)NHH*DH*DH);
  __bf16* vw = WB(ar, st, p.v, (long)NHH*DH*DH);
  __bf16 *qb = ar.bf((size_t)BBATCH*SI), *kb = ar.bf((size_t)BBATCH*SI), *vb = ar.bf((size_t)BBATCH*SI);
  for (int n = 0; n < NHH; ++n) {
    GEMM<EPI_NONE,false,false,true>(st, ca+n*DH,SI,I, qw+(long)n*DH*DH,0,DH, 0,0, 0,0,0, qb+n*DH,SI,I, S,DH,DH);
    GEMM<EPI_NONE,false,false,true>(st, ca+n*DH,SI,I, kw+(long)n*DH*DH,0,DH, 0,0, 0,0,0, kb+n*DH,SI,I, S,DH,DH);
    GEMM<EPI_NONE,false,false,true>(st, xm+n*DH,SI,I, vw+(long)n*DH*DH,0,DH, 0,0, 0,0,0, vb+n*DH,SI,I, S,DH,DH);
  }
  __bf16* ifin = ar.bf((size_t)BBATCH*S*K3);
  k_cat3<<<g1((long)BBATCH*S*K3), 256, 0, st>>>(qb, kb, vb, ifin, (long)BBATCH*S, I);

  __bf16* igw = WB(ar, st, p.igw, (long)NHH*K3);
  __bf16* fgw = WB(ar, st, p.fgw, (long)NHH*K3);
  float* igf = ar.f32((size_t)BBATCH*S*NHH);
  float* fgf = ar.f32((size_t)BBATCH*S*NHH);
  GEMM<EPI_BIAS,true,false,false>(st, ifin,(long)S*K3,K3, igw,0,K3, 0,p.igb, 0,0,0, igf,(long)S*NHH,NHH, S,NHH,K3);
  GEMM<EPI_BIAS,true,false,false>(st, ifin,(long)S*K3,K3, fgw,0,K3, 0,p.fgb, 0,0,0, fgf,(long)S*NHH,NHH, S,NHH,K3);

  float* cum = ar.f32((size_t)BBATCH*NHH*S);
  float* igT = ar.f32((size_t)BBATCH*NHH*S);
  k_cumgates<<<dim3(NHH, BBATCH), 32, 0, st>>>(igf, fgf, cum, igT, S);

  const long NP = (long)BBATCH*NHH*S*DHP;
  __bf16 *qp = ar.bf((size_t)NP), *kp = ar.bf((size_t)NP), *vp = ar.bf((size_t)NP);
  k_pack_pad<<<g1(NP), 256, 0, st>>>(qb, qp, S, DH, DHP);
  k_pack_pad<<<g1(NP), 256, 0, st>>>(kb, kp, S, DH, DHP);
  k_pack_pad_T<<<g1(NP), 256, 0, st>>>(vb, vp, S, DH, DHP);   // V transposed

  float* Hf = ar.f32((size_t)NP);
  const float sc = 1.f / sqrtf((float)DH);
  dim3 mg((S + 63) / 64, NHH, BBATCH);
  switch (DHP) {
    case 32:  k_mlstm<32><<<mg,128,0,st>>>(qp,kp,vp,cum,igT,Hf,S,sc); break;
    case 64:  k_mlstm<64><<<mg,128,0,st>>>(qp,kp,vp,cum,igT,Hf,S,sc); break;
    case 96:  k_mlstm<96><<<mg,128,0,st>>>(qp,kp,vp,cum,igT,Hf,S,sc); break;
    default:  k_mlstm<192><<<mg,128,0,st>>>(qp,kp,vp,cum,igT,Hf,S,sc); break;
  }

  __bf16* hn = ar.bf((size_t)BBATCH*SI);
  k_headln<<<g1((long)BBATCH*NHH*S), 256, 0, st>>>(Hf, p.mh_g, hn, S, DH, DHP);
  __bf16* hg = ar.bf((size_t)BBATCH*SI);
  k_hs_gate<<<g1((long)BBATCH*SI), 256, 0, st>>>(hn, ca, z, p.skip, hg, (long)BBATCH*S, I);

  __bf16* downW = WB(ar, st, p.down, (long)dim*I);
  __bf16* xo = ar.bf((size_t)BBATCH*SD);
  GEMM<EPI_NONE,false,true,true>(st, hg,SI,I, downW,0,I, 0,0, x,SD,dim, xo,SD,dim, S,dim,I);
  return xo;
}

static __bf16* run_ds(hipStream_t st, Arena& ar, const DsP& p,
                      const __bf16* x, int H, int W, int cin, int cout) {
  const long S = (long)H * W;
  __bf16* dwo = ar.bf((size_t)BBATCH*S*cin);
  k_dw3x3<<<g1((long)BBATCH*S*cin), 256, 0, st>>>(x, p.dw, p.dwb, dwo, H, W, cin);
  float* s  = ar.f32(cout);
  float* b2 = ar.f32(cout);
  k_fold_bn<<<g1(cout), 256, 0, st>>>(p.g, p.b, p.pwb, s, b2, cout);
  __bf16* pw = WB(ar, st, p.pw, (long)cout*cin);
  __bf16* o  = ar.bf((size_t)BBATCH*S*cout);
  GEMM<EPI_SB_RELU,false,false,true>(st, dwo,S*cin,cin, pw,0,cin, s,b2, 0,0,0, o,S*cout,cout, (int)S,cout,cin);
  return o;
}

static __bf16* run_mlp(hipStream_t st, Arena& ar, const MlpP& p,
                       const __bf16* x, long S, int cin, int cout, const __bf16* res) {
  __bf16* w1 = WB(ar, st, p.w1, (long)cin*cin);
  __bf16* w2 = WB(ar, st, p.w2, (long)cout*cin);
  __bf16* h = ar.bf((size_t)BBATCH*S*cin);
  GEMM<EPI_BIAS_GELU,false,false,true>(st, x,S*cin,cin, w1,0,cin, 0,p.b1, 0,0,0, h,S*cin,cin, (int)S,cin,cin);
  __bf16* o = ar.bf((size_t)BBATCH*S*cout);
  if (res)
    GEMM<EPI_BIAS,false,true,true>(st, h,S*cin,cin, w2,0,cin, 0,p.b2, res,S*cout,cout, o,S*cout,cout, (int)S,cout,cin);
  else
    GEMM<EPI_BIAS,false,false,true>(st, h,S*cin,cin, w2,0,cin, 0,p.b2, 0,0,0, o,S*cout,cout, (int)S,cout,cin);
  return o;
}

static __bf16* bidir_vil(hipStream_t st, Arena& ar, const VilP& p,
                         const __bf16* d, int S, int C) {
  __bf16* v1 = run_vil(st, ar, p, d, S, C);
  __bf16* df = ar.bf((size_t)BBATCH*S*C);
  k_flip<<<g1((long)BBATCH*S*C), 256, 0, st>>>(d, df, S, C);
  __bf16* v2 = run_vil(st, ar, p, df, S, C);
  __bf16* xl = ar.bf((size_t)BBATCH*S*C);
  k_addflip<<<g1((long)BBATCH*S*C), 256, 0, st>>>(v1, v2, xl, S, C);
  return xl;
}

static __bf16* gate_branch(hipStream_t st, Arena& ar, const DsP& ds, const PcP& pc,
                           const __bf16* d, const __bf16* xcl,
                           int H, int W, int C) {
  const long S = (long)H * W;
  __bf16* cat = ar.bf((size_t)BBATCH*S*2*C);
  k_cat2<<<g1((long)BBATCH*S*2*C), 256, 0, st>>>(d, xcl, cat, (long)BBATCH*S, C, C);
  __bf16* dd = run_ds(st, ar, ds, cat, H, W, 2*C, C);
  __bf16* w = WB(ar, st, pc.w, C);
  __bf16* aw = ar.bf((size_t)BBATCH*S);
  GEMM<EPI_BIAS_SIGMOID,false,false,false>(st, dd,S*C,C, w,0,C, 0,pc.b, 0,0,0, aw,S,1, (int)S,1,C);
  return aw;
}

static __bf16* run_atten(hipStream_t st, Arena& ar, const AttenP& p,
                         const float* fA, const float* fB, int H, int W, int C) {
  const long S = (long)H * W;
  const long n = (long)BBATCH * S * C;
  __bf16 *d = ar.bf((size_t)n), *acl = ar.bf((size_t)n), *bcl = ar.bf((size_t)n);
  k_tobf_cl<<<g1(n), 256, 0, st>>>(fA, fB, d, C, S, 1);
  k_tobf_cl<<<g1(n), 256, 0, st>>>(fA, fA, acl, C, S, 0);
  k_tobf_cl<<<g1(n), 256, 0, st>>>(fB, fB, bcl, C, S, 0);
  __bf16* xl = bidir_vil(st, ar, p.v, d, (int)S, C);
  __bf16* awA = gate_branch(st, ar, p.catA, p.convA, d, acl, H, W, C);
  __bf16* awB = gate_branch(st, ar, p.catB, p.convB, d, bcl, H, W, C);
  __bf16 *t1 = ar.bf((size_t)n), *t2 = ar.bf((size_t)n);
  k_gatemul<<<g1(n), 256, 0, st>>>(awA, nullptr, xl, t1, S, C, 0);
  k_gatemul<<<g1(n), 256, 0, st>>>(awB, nullptr, xl, t2, S, C, 0);
  __bf16* cc = ar.bf((size_t)BBATCH*S*2*C);
  k_cat2<<<g1((long)BBATCH*S*2*C), 256, 0, st>>>(t1, t2, cc, (long)BBATCH*S, C, C);
  return run_ds(st, ar, p.cat, cc, H, W, 2*C, C);
}

static __bf16* run_axial(hipStream_t st, Arena& ar, const AxialP& p,
                         const float* fA, const float* fB, int H, int W, int C) {
  const long S = (long)H * W;
  const long n = (long)BBATCH * S * C;
  __bf16 *d = ar.bf((size_t)n), *acl = ar.bf((size_t)n), *bcl = ar.bf((size_t)n);
  k_tobf_cl<<<g1(n), 256, 0, st>>>(fA, fB, d, C, S, 1);
  k_tobf_cl<<<g1(n), 256, 0, st>>>(fA, fA, acl, C, S, 0);
  k_tobf_cl<<<g1(n), 256, 0, st>>>(fB, fB, bcl, C, S, 0);

  __bf16* hm = ar.bf((size_t)BBATCH*H*C);
  __bf16* wm = ar.bf((size_t)BBATCH*W*C);
  k_mean<<<g1((long)BBATCH*H*C), 256, 0, st>>>(d, hm, H, W, C, 1);
  k_mean<<<g1((long)BBATCH*W*C), 256, 0, st>>>(d, wm, H, W, C, 0);
  k_pos_add<<<g1((long)BBATCH*H*C), 256, 0, st>>>(hm, p.ph, H, C);
  k_pos_add<<<g1((long)BBATCH*W*C), 256, 0, st>>>(wm, p.pw, W, C);

  __bf16* xh = bidir_vil(st, ar, p.vh, hm, H, C);
  __bf16* xw = bidir_vil(st, ar, p.vw, wm, W, C);

  __bf16* comb = ar.bf((size_t)n);
  k_comb3<<<g1(n), 256, 0, st>>>(d, xh, xw, comb, H, W, C);
  float* s  = ar.f32(C);
  float* b2 = ar.f32(C);
  k_fold_bn<<<g1(C), 256, 0, st>>>(p.xc.g, p.xc.b, nullptr, s, b2, C);
  __bf16* xcw = WB(ar, st, p.xc.w, (long)C*C);
  __bf16* xx = ar.bf((size_t)n);
  GEMM<EPI_SB_RELU_SIG,false,false,true>(st, comb,S*C,C, xcw,0,C, s,b2, 0,0,0, xx,S*C,C, (int)S,C,C);

  __bf16* awA = gate_branch(st, ar, p.catA, p.convA, d, acl, H, W, C);
  __bf16* awB = gate_branch(st, ar, p.catB, p.convB, d, bcl, H, W, C);
  __bf16 *t1 = ar.bf((size_t)n), *t2 = ar.bf((size_t)n);
  k_gatemul<<<g1(n), 256, 0, st>>>(awA, xx, acl, t1, S, C, 1);
  k_gatemul<<<g1(n), 256, 0, st>>>(awB, xx, bcl, t2, S, C, 1);
  __bf16* cc = ar.bf((size_t)BBATCH*S*2*C);
  k_cat2<<<g1((long)BBATCH*S*2*C), 256, 0, st>>>(t1, t2, cc, (long)BBATCH*S, C, C);
  return run_ds(st, ar, p.cat, cc, H, W, 2*C, C);
}

static __bf16* run_lh(hipStream_t st, Arena& ar, const LhP& p,
                      const __bf16* xl, int Hl, int Wl, int cl,
                      const __bf16* xh, int Hh, int Wh, int ch) {
  const long Sl = (long)Hl * Wl, Sh = (long)Hh * Wh;
  __bf16* xl2 = run_mlp(st, ar, p.mlp_l, xl, Sl, cl, cl, xl);

  float *cs = ar.f32(ch), *cb2 = ar.f32(ch);
  k_fold_bn<<<g1(ch), 256, 0, st>>>(p.conv.g, p.conv.b, nullptr, cs, cb2, ch);
  __bf16* cw = WB(ar, st, p.conv.w, (long)ch*cl);
  __bf16* cup = ar.bf((size_t)BBATCH*Sl*ch);
  GEMM<EPI_SB_RELU,false,false,true>(st, xl2,Sl*cl,cl, cw,0,cl, cs,cb2, 0,0,0, cup,Sl*ch,ch, (int)Sl,ch,cl);
  __bf16* upb = ar.bf((size_t)BBATCH*Sh*ch);
  k_bilinear<<<g1((long)BBATCH*Sh*ch), 256, 0, st>>>(cup, upb, Hl, Wl, Hh, Wh, ch);
  __bf16* xh2 = ar.bf((size_t)BBATCH*Sh*ch);
  k_add2<<<g1((long)BBATCH*Sh*ch), 256, 0, st>>>(xh, upb, xh2, (long)BBATCH*Sh*ch);

  float *bs = ar.f32(cl), *bb = ar.f32(cl);
  k_fold_bn<<<g1(cl), 256, 0, st>>>(p.bn_g, p.bn_b, nullptr, bs, bb, cl);
  __bf16* bnl = ar.bf((size_t)BBATCH*Sl*cl);
  k_affine<<<g1((long)BBATCH*Sl*cl), 256, 0, st>>>(xl2, bs, bb, bnl, (long)BBATCH*Sl, cl);
  __bf16* pooled = ar.bf((size_t)BBATCH*144*cl);
  k_amp12<<<g1((long)BBATCH*144*cl), 256, 0, st>>>(bnl, pooled, Hl, Wl, cl);

  __bf16* kvw = WB(ar, st, p.kv.w, (long)2*ch*cl);
  __bf16* kv = ar.bf((size_t)BBATCH*144*2*ch);
  GEMM<EPI_BIAS,false,false,true>(st, pooled,(long)144*cl,cl, kvw,0,cl, 0,p.kv.b, 0,0,0,
                                  kv,(long)144*2*ch,2*ch, 144,2*ch,cl);

  float *ns = ar.f32(ch), *nb2 = ar.f32(ch);
  k_fold_bn<<<g1(ch), 256, 0, st>>>(p.ng, p.nb, nullptr, ns, nb2, ch);
  __bf16* xn = ar.bf((size_t)BBATCH*Sh*ch);
  k_affine<<<g1((long)BBATCH*Sh*ch), 256, 0, st>>>(xh2, ns, nb2, xn, (long)BBATCH*Sh, ch);

  float* logit = ar.f32((size_t)BBATCH*Sh*144);
  GEMM<EPI_NONE,true,false,true>(st, xn,Sh*ch,ch, kv,(long)144*2*ch,2*ch, 0,0, 0,0,0,
                                 logit,Sh*144,144, (int)Sh,144,ch);
  __bf16* att = ar.bf((size_t)BBATCH*Sh*144);
  k_softmax_row<<<g1((long)BBATCH*Sh), 256, 0, st>>>(logit, att, (long)BBATCH*Sh, 144, 1.f/12.f);

  __bf16* vmt = ar.bf((size_t)BBATCH*ch*144);
  k_transpose_bf<<<g1((long)BBATCH*ch*144), 256, 0, st>>>(kv, vmt, 144, ch, 2*ch, ch);
  __bf16* xh3 = ar.bf((size_t)BBATCH*Sh*ch);
  GEMM<EPI_NONE,false,true,true>(st, att,Sh*144,144, vmt,(long)ch*144,144, 0,0,
                                 xh2,Sh*ch,ch, xh3,Sh*ch,ch, (int)Sh,ch,144);
  return run_mlp(st, ar, p.mlp_h, xh3, Sh, ch, ch, xh3);
}

// ---------------------------------------------------------------------------
extern "C" void kernel_launch(void* const* d_in, const int* in_sizes, int n_in,
                              void* d_out, int out_size, void* d_ws, size_t ws_size,
                              hipStream_t stream) {
  (void)in_sizes; (void)out_size;
  Arena ar{(char*)d_ws, ws_size, 0};
  PCur P{d_in, 8, nullptr, 0, false};
  if (n_in == 9) { P.blobMode = true; P.blob = (const float*)d_in[8]; }

  // pytree (alphabetical) parameter order
  DsP    dwc  = takeDs(P, 40, 40);
  AxialP f0   = takeAxial(P, 40);
  AxialP f1   = takeAxial(P, 80);
  AttenP f2   = takeAtten(P, 192);
  AttenP f3   = takeAtten(P, 384);
  LhP    lh1  = takeLh(P, 80, 40);
  LhP    lh2  = takeLh(P, 192, 40);
  LhP    lh3  = takeLh(P, 384, 40);
  MlpP   mlp1 = takeMlp(P, 40, 40);
  MlpP   mlp2 = takeMlp(P, 40, 2);

  const float* fa[4] = {(const float*)d_in[0], (const float*)d_in[1],
                        (const float*)d_in[2], (const float*)d_in[3]};
  const float* fb[4] = {(const float*)d_in[4], (const float*)d_in[5],
                        (const float*)d_in[6], (const float*)d_in[7]};

  __bf16* x0 = run_axial(stream, ar, f0, fa[0], fb[0], 128, 128, 40);
  __bf16* x1 = run_axial(stream, ar, f1, fa[1], fb[1], 64, 64, 80);
  __bf16* x2 = run_atten(stream, ar, f2, fa[2], fb[2], 32, 32, 192);
  __bf16* x3 = run_atten(stream, ar, f3, fa[3], fb[3], 16, 16, 384);

  __bf16* xh = run_lh(stream, ar, lh1, x1, 64, 64, 80, x0, 128, 128, 40);
  xh = run_lh(stream, ar, lh2, x2, 32, 32, 192, xh, 128, 128, 40);
  xh = run_lh(stream, ar, lh3, x3, 16, 16, 384, xh, 128, 128, 40);

  const long S0 = 128L * 128L;
  __bf16* m1 = run_mlp(stream, ar, mlp1, xh, S0, 40, 40, nullptr);
  __bf16* dd = run_ds(stream, ar, dwc, xh, 128, 128, 40, 40);
  __bf16* sum = ar.bf((size_t)BBATCH * S0 * 40);
  k_add2<<<g1((long)BBATCH*S0*40), 256, 0, stream>>>(dd, m1, sum, (long)BBATCH*S0*40);
  __bf16* o2 = run_mlp(stream, ar, mlp2, sum, S0, 40, 2, nullptr);

  __bf16* up = ar.bf((size_t)BBATCH * 512 * 512 * 2);
  k_bilinear<<<g1((long)BBATCH*512*512*2), 256, 0, stream>>>(o2, up, 128, 128, 512, 512, 2);
  k_slc2nchw<<<g1((long)BBATCH*2*512*512), 256, 0, stream>>>(up, (float*)d_out, 512, 512, 2);
}
